// MatrixAttentionLayer_39788577030271
// MI455X (gfx1250) — matrix operand order repaired and checked by reading, not compiled, not run on hardware
//
#include <hip/hip_runtime.h>

#define DEVINL __device__ __forceinline__

typedef __attribute__((ext_vector_type(16))) __bf16        v16bf;
typedef __attribute__((ext_vector_type(8)))  float         v8f;
typedef __attribute__((ext_vector_type(8)))  unsigned int  v8u;

constexpr int B = 8, S = 4096, T = 512, H = 768;

// float -> bf16 bits, round-to-nearest-even
DEVINL unsigned short f2bf(float f) {
  unsigned u = __builtin_bit_cast(unsigned, f);
  u += 0x7FFFu + ((u >> 16) & 1u);
  return (unsigned short)(u >> 16);
}

DEVINL v16bf make_frag(uint4 a, uint4 b) {
  v8u r;
  r[0] = a.x; r[1] = a.y; r[2] = a.z; r[3] = a.w;
  r[4] = b.x; r[5] = b.y; r[6] = b.z; r[7] = b.w;
  return __builtin_bit_cast(v16bf, r);
}

// A fragment (16xK tile, row-major bf16 bits, row stride ldk elements).
// ISA layout: lanes 0-15 hold M=lane, K = k0+[0..7] and k0+[16..23];
//             lanes 16-31 hold M=lane-16, K = k0+[8..15] and k0+[24..31].
DEVINL v16bf load_a(const unsigned short* base, int ldk, int k0, int lane) {
  int m  = lane & 15;
  int kb = k0 + ((lane & 16) ? 8 : 0);
  const uint4* p = (const uint4*)(base + (size_t)m * ldk + kb);
  const uint4* q = (const uint4*)(base + (size_t)m * ldk + kb + 16);
  return make_frag(*p, *q);
}

// B fragment: B[k][n] = rows[n0+n][k0+k] (rows row-major, stride ldk).
// ISA layout: lanes 0-15 N=lane with K=k0+[0..7],k0+[16..23]; lanes 16-31 N=lane-16 with K=k0+[8..15],k0+[24..31].
DEVINL v16bf load_b(const unsigned short* rows, int ldk, int n0, int k0, int lane) {
  int n  = lane & 15;
  int kb = k0 + ((lane & 16) ? 8 : 0);
  const uint4* p = (const uint4*)(rows + (size_t)(n0 + n) * ldk + kb);
  return make_frag(p[0], p[2]);
}

DEVINL v8f wmma_bf16(v16bf a, v16bf b, v8f c) {
  return __builtin_amdgcn_wmma_f32_16x16x32_bf16(false, a, false, b, (short)0, c, false, false);
}

// Software-pipelined K-loop: fragments for step k+1 are loaded before the
// WMMAs of step k are issued, so the pre-WMMA wait is only on loads that have
// had a full iteration of latency (partial s_wait_loadcnt instead of ==0).
template <int NT>
DEVINL void gemm_rows(const unsigned short* aLds, int aLdk, int Ktot,
                      const unsigned short* bRows, int bLdk, int n0, int lane, v8f* acc) {
  v16bf a_cur = load_a(aLds, aLdk, 0, lane);
  v16bf b_cur[NT];
#pragma unroll
  for (int t = 0; t < NT; ++t) b_cur[t] = load_b(bRows, bLdk, n0 + t * 16, 0, lane);
  for (int k0 = 32; k0 < Ktot; k0 += 32) {
    v16bf a_nxt = load_a(aLds, aLdk, k0, lane);
    v16bf b_nxt[NT];
#pragma unroll
    for (int t = 0; t < NT; ++t) b_nxt[t] = load_b(bRows, bLdk, n0 + t * 16, k0, lane);
#pragma unroll
    for (int t = 0; t < NT; ++t) acc[t] = wmma_bf16(a_cur, b_cur[t], acc[t]);
    a_cur = a_nxt;
#pragma unroll
    for (int t = 0; t < NT; ++t) b_cur[t] = b_nxt[t];
  }
#pragma unroll
  for (int t = 0; t < NT; ++t) acc[t] = wmma_bf16(a_cur, b_cur[t], acc[t]);
}

// ---------------------------------------------------------------- prep kernels
__global__ void __launch_bounds__(256) cvt_w_kernel(const float* __restrict__ Wf,
                                                    unsigned short* __restrict__ Wb) {
  int idx = blockIdx.x * 256 + threadIdx.x;
  if (idx < H * H) Wb[idx] = f2bf(Wf[idx]);
}

// tgtT[b][h][t] = target[b][t][h]  (bf16)
__global__ void __launch_bounds__(256) transpose_tgt_kernel(const float* __restrict__ tgt,
                                                            unsigned short* __restrict__ tT) {
  int idx = blockIdx.x * 256 + threadIdx.x;
  if (idx >= B * H * T) return;
  int t  = idx % T;
  int bh = idx / T;
  int h  = bh % H;
  int bb = bh / H;
  tT[idx] = f2bf(tgt[((size_t)bb * T + t) * H + h]);
}

// ------------------------------------------- Y = relu((X*mask) @ W^T) in bf16
// grid.x = nRows/16, block = 256 (8 waves, 6 col-tiles each -> 48 col tiles = H)
__global__ void __launch_bounds__(256) transform_kernel(const float* __restrict__ X,
                                                        const int* __restrict__ mask,
                                                        const unsigned short* __restrict__ Wb,
                                                        unsigned short* __restrict__ Y) {
  __shared__ __align__(16) unsigned short aTile[16 * H];  // 24 KB
  const int row0 = blockIdx.x * 16;
  const size_t xbase = (size_t)row0 * H;
  for (int idx = threadIdx.x; idx < 16 * H; idx += 256) {
    int r = idx / H;
    float m = (float)mask[row0 + r];
    aTile[idx] = f2bf(X[xbase + idx] * m);
  }
  __syncthreads();

  const int lane = threadIdx.x & 31;
  const int w    = threadIdx.x >> 5;
  v8f acc[6] = {};
  gemm_rows<6>(aTile, H, H, Wb, H, w * 96, lane, acc);

  const int n  = lane & 15;
  const int mb = (lane & 16) ? 8 : 0;
#pragma unroll
  for (int t = 0; t < 6; ++t) {
    int h = w * 96 + t * 16 + n;
#pragma unroll
    for (int v = 0; v < 8; ++v) {
      float x = acc[t][v];
      x = x > 0.f ? x : 0.f;  // relu
      Y[(size_t)(row0 + mb + v) * H + h] = f2bf(x);
    }
  }
}

// -------- attention: scores = seq_t @ tgt_t^T ; softmax over T ; @ target
// grid = (S/16, B), block = 256 (8 waves)
__global__ void __launch_bounds__(256) attn_kernel(const unsigned short* __restrict__ seq_t,
                                                   const unsigned short* __restrict__ tgt_t,
                                                   const unsigned short* __restrict__ tgtT,
                                                   const int* __restrict__ seq_mask,
                                                   float* __restrict__ cat_out,
                                                   float* __restrict__ attn_out) {
  __shared__ __align__(16) float scoresBuf[16 * T];        // 32 KB (phase0/1: seq_t staging)
  __shared__ __align__(16) unsigned short attnP[16 * T];   // 16 KB
  __shared__ float invSum[16];
  __shared__ float smRow[16];

  const int b    = blockIdx.y;
  const int s0   = blockIdx.x * 16;
  const int lane = threadIdx.x & 31;
  const int w    = threadIdx.x >> 5;

  // Phase 0: stage 16 rows of seq_t (bf16, contiguous) + row masks
  unsigned short* aTile = (unsigned short*)scoresBuf;  // 24 KB of the 32 KB buffer
  {
    const size_t base = ((size_t)b * S + s0) * H;
    for (int idx = threadIdx.x; idx < 16 * H; idx += 256) aTile[idx] = seq_t[base + idx];
    if (threadIdx.x < 16) smRow[threadIdx.x] = (float)seq_mask[b * S + s0 + threadIdx.x];
  }
  __syncthreads();

  // Phase 1: scores; wave w covers T columns [w*64, w*64+64)
  v8f sc[4] = {};
  gemm_rows<4>(aTile, H, H, tgt_t + (size_t)b * T * H, H, w * 64, lane, sc);

  __syncthreads();  // aTile reads done -> reuse buffer for f32 scores
  {
    const int n  = lane & 15;
    const int mb = (lane & 16) ? 8 : 0;
#pragma unroll
    for (int t = 0; t < 4; ++t) {
      int c = w * 64 + t * 16 + n;
#pragma unroll
      for (int v = 0; v < 8; ++v) scoresBuf[(mb + v) * T + c] = sc[t][v];
    }
  }
  __syncthreads();

  // Phase 2: softmax over T=512; wave w handles rows 2w, 2w+1
#pragma unroll
  for (int rr = 0; rr < 2; ++rr) {
    int r = w * 2 + rr;
    float mx = -3.0e38f;
    for (int c = lane; c < T; c += 32) mx = fmaxf(mx, scoresBuf[r * T + c]);
#pragma unroll
    for (int off = 16; off > 0; off >>= 1) mx = fmaxf(mx, __shfl_xor(mx, off, 32));
    float sum = 0.f;
    for (int c = lane; c < T; c += 32) {
      float p = __expf(scoresBuf[r * T + c] - mx);
      sum += p;
      attnP[r * T + c] = f2bf(p);  // unnormalized; 1/sum folded into output scale
    }
#pragma unroll
    for (int off = 16; off > 0; off >>= 1) sum += __shfl_xor(sum, off, 32);
    if (lane == 0) invSum[r] = 1.0f / sum;
  }
  __syncthreads();

  // Phase 3: attn_out = attnP @ target ; wave w covers H columns [w*96, +96)
  v8f ao[6] = {};
  gemm_rows<6>(attnP, T, T, tgtT + (size_t)b * H * T, T, w * 96, lane, ao);
  {
    const int n  = lane & 15;
    const int mb = (lane & 16) ? 8 : 0;
#pragma unroll
    for (int t = 0; t < 6; ++t) {
      int h = w * 96 + t * 16 + n;
#pragma unroll
      for (int v = 0; v < 8; ++v) {
        int m = mb + v;
        float val = ao[t][v] * invSum[m];
        size_t srow = (size_t)b * S + (s0 + m);
        attn_out[srow * H + h]          = val;
        cat_out[srow * (2 * H) + H + h] = val * smRow[m];
      }
    }
  }
}

// cat_out[:, :, 0:H] = seq_output * seq_mask
__global__ void __launch_bounds__(256) cat_seq_kernel(const float* __restrict__ seq,
                                                      const int* __restrict__ mask,
                                                      float* __restrict__ cat_out) {
  size_t idx = (size_t)blockIdx.x * 256 + threadIdx.x;
  if (idx >= (size_t)B * S * H) return;
  size_t bs = idx / H;
  int h = (int)(idx - bs * H);
  cat_out[bs * (2 * H) + h] = seq[idx] * (float)mask[bs];
}

extern "C" void kernel_launch(void* const* d_in, const int* in_sizes, int n_in,
                              void* d_out, int out_size, void* d_ws, size_t ws_size,
                              hipStream_t stream) {
  (void)in_sizes; (void)n_in; (void)out_size; (void)ws_size;
  const float* seq   = (const float*)d_in[0];
  const int*   smask = (const int*)d_in[1];
  const float* tgt   = (const float*)d_in[2];
  const int*   tmask = (const int*)d_in[3];
  const float* Wf    = (const float*)d_in[4];

  char* ws = (char*)d_ws;
  unsigned short* Wb    = (unsigned short*)ws;  ws += (size_t)H * H * 2;      // 1.2 MB
  unsigned short* seqT  = (unsigned short*)ws;  ws += (size_t)B * S * H * 2;  // 50.3 MB
  unsigned short* tgtTf = (unsigned short*)ws;  ws += (size_t)B * T * H * 2;  // 6.3 MB (tgt_t)
  unsigned short* tgtTr = (unsigned short*)ws;  ws += (size_t)B * H * T * 2;  // 6.3 MB (target^T)

  float* cat_out  = (float*)d_out;
  float* attn_out = cat_out + (size_t)B * S * 2 * H;

  hipLaunchKernelGGL(cvt_w_kernel, dim3((H * H + 255) / 256), dim3(256), 0, stream, Wf, Wb);
  hipLaunchKernelGGL(transpose_tgt_kernel, dim3((B * H * T + 255) / 256), dim3(256), 0, stream,
                     tgt, tgtTr);
  hipLaunchKernelGGL(transform_kernel, dim3(B * S / 16), dim3(256), 0, stream,
                     seq, smask, Wb, seqT);
  hipLaunchKernelGGL(transform_kernel, dim3(B * T / 16), dim3(256), 0, stream,
                     tgt, tmask, Wb, tgtTf);
  hipLaunchKernelGGL(attn_kernel, dim3(S / 16, B), dim3(256), 0, stream,
                     seqT, tgtTf, tgtTr, smask, cat_out, attn_out);
  hipLaunchKernelGGL(cat_seq_kernel, dim3((int)(((size_t)B * S * H + 255) / 256)), dim3(256), 0,
                     stream, seq, smask, cat_out);
}
